// SlidingWindowAttention_34892314312809
// MI455X (gfx1250) — matrix operand order repaired and checked by reading, not compiled, not run on hardware
//
#include <hip/hip_runtime.h>
#include <hip/hip_bf16.h>

// ---------------- problem constants ----------------
constexpr int B_    = 2;
constexpr int S_    = 2048;
constexpr int E_    = 1024;
constexpr int H_    = 16;
constexpr int KV_   = 4;
constexpr int D_    = 64;        // head dim
constexpr int NREP_ = H_ / KV_;  // 4
constexpr int WIN_  = 256;
constexpr int KVDIM_ = KV_ * D_; // 256
constexpr int BS_   = B_ * S_;   // 4096 tokens

constexpr int LDSS = 40;         // LDS row stride in halves (80 B: 16B-aligned, conflict-breaking)

typedef __attribute__((ext_vector_type(16))) _Float16 v16h;
typedef __attribute__((ext_vector_type(8)))  _Float16 v8h;
typedef __attribute__((ext_vector_type(8)))  float    v8f;

// ---------------- WMMA helpers ----------------
__device__ __forceinline__ v8f wmma_f16(v16h a, v16h b, v8f c) {
  // D = A(16x32 f16) * B(32x16 f16) + C(16x16 f32)
  return __builtin_amdgcn_wmma_f32_16x16x32_f16(
      /*neg_a=*/false, a, /*neg_b=*/false, b,
      /*c_mod=*/(short)0, c, /*reuse_a=*/false, /*reuse_b=*/false);
}

__device__ __forceinline__ v16h cat8(v8h lo, v8h hi) {
  v16h r;
#pragma unroll
  for (int i = 0; i < 8; ++i) { r[i] = lo[i]; r[8 + i] = hi[i]; }
  return r;
}

// A-fragment (16x32, row-major source, row stride ldh in halves).
// Lane layout (ISA 7.12.2): lo lanes: K 0..7 then 16..23 ; hi lanes: K 8..15 then 24..31.
__device__ __forceinline__ v16h load_a_frag(const _Float16* p, int ldh) {
  const int lane = threadIdx.x & 31;
  const int rl = lane & 15, hi = lane >> 4;
  const _Float16* rp = p + (size_t)rl * ldh;
  v8h lo8 = *(const v8h*)(rp + 8 * hi);
  v8h hi8 = *(const v8h*)(rp + 16 + 8 * hi);
  return cat8(lo8, hi8);
}

// B-fragment (32x16) from a K-contiguous-per-N buffer: element (k,n) at p[n*ldh + k].
// Lane layout: lo lanes K 0..7 then 16..23 ; hi lanes K 8..15 then 24..31; n = lane&15.
__device__ __forceinline__ v16h load_b_frag_T(const _Float16* p, int ldh) {
  const int lane = threadIdx.x & 31;
  const int nl = lane & 15, hi = lane >> 4;
  const _Float16* rp = p + (size_t)nl * ldh + 8 * hi;
  return cat8(*(const v8h*)rp, *(const v8h*)(rp + 16));
}

// Async global->LDS copy of 32 bytes per lane (two b128 with shared addr regs;
// IOFFSET applies to both LDS and global addresses per CDNA5 ISA).
__device__ __forceinline__ void async_copy_b128x2(const void* gsrc, void* lds_dst) {
  unsigned lds_off = (unsigned)(uintptr_t)lds_dst;              // low 32 bits = LDS offset
  unsigned long long gaddr = (unsigned long long)(uintptr_t)gsrc;
  asm volatile("global_load_async_to_lds_b128 %0, %1, off"
               :: "v"(lds_off), "v"(gaddr) : "memory");
  asm volatile("global_load_async_to_lds_b128 %0, %1, off offset:16"
               :: "v"(lds_off), "v"(gaddr) : "memory");
}

__device__ __forceinline__ void wait_asynccnt0() {
  asm volatile("s_wait_asynccnt 0x0" ::: "memory");
}

// ---------------- f32 -> f16 conversion ----------------
__global__ void cvt_f32_f16_kernel(const float* __restrict__ in,
                                   _Float16* __restrict__ out, int n) {
  int i = blockIdx.x * blockDim.x + threadIdx.x;
  int stride = gridDim.x * blockDim.x;
  for (; i < n; i += stride) out[i] = (_Float16)in[i];
}

// ---------------- tiled WMMA GEMM: C[M,N] = A[M,K] * W[K,N] + bias ----------------
// Block: 256 threads = 8 waves (4 in M x 2 in N); block tile 128(M) x 64(N);
// each wave owns a 32x32 tile = 2x2 WMMA accumulators. K staged in 32-chunks via LDS.
// A-tile goes through the CDNA5 async global->LDS engine (ASYNCcnt); W-tile is
// transposed through registers into LDS so B-fragments are contiguous ds reads.
template <bool F16OUT>
__global__ void gemm_wmma_kernel(const _Float16* __restrict__ A,
                                 const _Float16* __restrict__ W,
                                 const float* __restrict__ bias,
                                 void* __restrict__ outp,
                                 int M, int N, int K) {
  __shared__ __align__(16) _Float16 lA[128 * LDSS]; // [row][k], 128 x 32 used
  __shared__ __align__(16) _Float16 lW[64 * LDSS];  // [n][k] transposed, 64 x 32 used

  const int tid  = threadIdx.x;
  const int lane = tid & 31;
  const int wave = tid >> 5;
  const int wm = wave & 3;   // wave row   (0..3)
  const int wn = wave >> 2;  // wave col   (0..1)
  const int m0 = blockIdx.x * 128;
  const int n0 = blockIdx.y * 64;

  v8f acc[2][2] = {};

  for (int kt = 0; kt < K; kt += 32) {
    // ---- stage A tile 128x32 via async global->LDS (2 threads per row) ----
    {
      int r = tid >> 1;
      int c = (tid & 1) * 16;
      const _Float16* src = A + (size_t)(m0 + r) * K + kt + c;
      async_copy_b128x2(src, &lA[r * LDSS + c]);
    }
    // ---- stage W tile 32x64, transposed into lW[n][k] ----
    {
      int k  = tid >> 3;         // 0..31
      int nb = (tid & 7) * 8;    // 0..56 step 8
      const _Float16* src = W + (size_t)(kt + k) * N + n0 + nb;
      v8h x = *(const v8h*)src;
#pragma unroll
      for (int j = 0; j < 8; ++j) lW[(nb + j) * LDSS + k] = x[j];
      // prefetch next k-step of the weight stream into cache
      if (kt + 32 < K) __builtin_prefetch(src + (size_t)32 * N, 0, 0);
    }
    wait_asynccnt0();
    __syncthreads();

    v16h af[2], bf[2];
#pragma unroll
    for (int i = 0; i < 2; ++i)
      af[i] = load_a_frag(&lA[(wm * 32 + i * 16) * LDSS], LDSS);
#pragma unroll
    for (int j = 0; j < 2; ++j)
      bf[j] = load_b_frag_T(&lW[(wn * 32 + j * 16) * LDSS], LDSS);
#pragma unroll
    for (int i = 0; i < 2; ++i)
#pragma unroll
      for (int j = 0; j < 2; ++j)
        acc[i][j] = wmma_f16(af[i], bf[j], acc[i][j]);
    __syncthreads();
  }

  // ---- epilogue: C/D layout -> memory; VGPR r holds rows r (lo) / r+8 (hi), col = lane&15 ----
  const int rl = lane & 15, hi = lane >> 4;
#pragma unroll
  for (int i = 0; i < 2; ++i) {
#pragma unroll
    for (int j = 0; j < 2; ++j) {
      int col = n0 + wn * 32 + j * 16 + rl;
      float bcol = bias[col];
#pragma unroll
      for (int r = 0; r < 8; ++r) {
        int row = m0 + wm * 32 + i * 16 + r + 8 * hi;
        float v = acc[i][j][r] + bcol;
        if (F16OUT) ((_Float16*)outp)[(size_t)row * N + col] = (_Float16)v;
        else        ((float*)outp)[(size_t)row * N + col]    = v;
      }
    }
  }
}

// ---------------- sliding-window flash attention ----------------
// One wave (32 threads) per (b, h, 16-query tile). Streams 32-key blocks over
// the causal window [i-256, i]. Q/K fragments come straight from global
// (contiguous per lane); V is transposed via LDS so PV B-fragments are
// contiguous ds reads; P is transposed C-layout->A-layout via LDS.
__global__ void attn_swa_kernel(const _Float16* __restrict__ qh,
                                const _Float16* __restrict__ kh,
                                const _Float16* __restrict__ vh,
                                _Float16* __restrict__ ch) {
  __shared__ __align__(16) _Float16 vt[64 * LDSS]; // [d][key] 64 x 32 used
  __shared__ __align__(16) _Float16 pb[16 * LDSS]; // [row][k] 16 x 32 used

  const int lane = threadIdx.x & 31;
  const int rl = lane & 15, hi = lane >> 4;

  int bid = blockIdx.x;
  const int qt = bid % (S_ / 16); bid /= (S_ / 16);
  const int h  = bid % H_;        const int b = bid / H_;
  const int q0 = qt * 16;
  const int kv = h / NREP_;

  // Q fragments: 16 rows x 64 d  ->  two 16x32 A-fragments
  const _Float16* qp = qh + (size_t)(b * S_ + q0) * E_ + h * D_;
  v16h qa[2];
#pragma unroll
  for (int c = 0; c < 2; ++c) qa[c] = load_a_frag(qp + 32 * c, E_);

  float m[8], l[8];
  v8f o[4] = {};
#pragma unroll
  for (int r = 0; r < 8; ++r) { m[r] = -1e30f; l[r] = 0.f; }

  int jb_first = q0 - WIN_; if (jb_first < 0) jb_first = 0;
  jb_first &= ~31;
  const int jb_last = (q0 + 15) & ~31;

  for (int jb = jb_first; jb <= jb_last; jb += 32) {
    // prefetch next key block of K into cache
    if (jb + 32 <= jb_last)
      __builtin_prefetch(kh + (size_t)(b * S_ + jb + 32 + lane) * KVDIM_ + kv * D_, 0, 0);

    // ---- stage V block transposed: vt[d][key]; each lane owns one key row ----
    {
      const _Float16* vp = vh + (size_t)(b * S_ + jb + lane) * KVDIM_ + kv * D_;
#pragma unroll
      for (int d8 = 0; d8 < 8; ++d8) {
        v8h x = *(const v8h*)(vp + d8 * 8);
#pragma unroll
        for (int j = 0; j < 8; ++j) vt[(d8 * 8 + j) * LDSS + lane] = x[j];
      }
    }

    // ---- S = Q K^T  (two 16-key half-tiles, two 32-d chunks each) ----
    v8f sc[2];
#pragma unroll
    for (int t = 0; t < 2; ++t) {
      v8f s = {};
#pragma unroll
      for (int c = 0; c < 2; ++c) {
        v16h kb = load_b_frag_T(
            kh + (size_t)(b * S_ + jb + 16 * t) * KVDIM_ + kv * D_ + 32 * c, KVDIM_);
        s = wmma_f16(qa[c], kb, s);
      }
      sc[t] = s;
    }

    // ---- scale + causal sliding-window mask ----
#pragma unroll
    for (int t = 0; t < 2; ++t) {
      int j = jb + 16 * t + rl;
#pragma unroll
      for (int r = 0; r < 8; ++r) {
        int i = q0 + r + 8 * hi;
        float v = sc[t][r] * 0.125f; // D^-0.5
        if (j > i || (i - j) > WIN_) v = -1e30f;
        sc[t][r] = v;
      }
    }

    // ---- online softmax (row stats across 16-lane half-groups) ----
    float pexp[2][8];
#pragma unroll
    for (int r = 0; r < 8; ++r) {
      float pm = fmaxf(sc[0][r], sc[1][r]);
#pragma unroll
      for (int msk = 1; msk < 16; msk <<= 1) pm = fmaxf(pm, __shfl_xor(pm, msk, 32));
      float mn = fmaxf(m[r], pm);
      float scale_old = __expf(m[r] - mn); // both -1e30 -> exp(0)=1, l stays 0
      float ps = 0.f;
#pragma unroll
      for (int t = 0; t < 2; ++t) {
        float p = (sc[t][r] <= -0.9e30f) ? 0.f : __expf(sc[t][r] - mn);
        pexp[t][r] = p;
        ps += p;
      }
#pragma unroll
      for (int msk = 1; msk < 16; msk <<= 1) ps += __shfl_xor(ps, msk, 32);
      l[r] = l[r] * scale_old + ps;
      m[r] = mn;
#pragma unroll
      for (int n = 0; n < 4; ++n) o[n][r] *= scale_old;
    }

    // ---- P: C-layout -> A-layout through LDS ----
    __syncthreads(); // also fences the vt stores above (single-wave block)
#pragma unroll
    for (int t = 0; t < 2; ++t)
#pragma unroll
      for (int r = 0; r < 8; ++r)
        pb[(r + 8 * hi) * LDSS + 16 * t + rl] = (_Float16)pexp[t][r];
    __syncthreads();

    v16h pa = load_a_frag(pb, LDSS);
#pragma unroll
    for (int n = 0; n < 4; ++n) {
      v16h vb = load_b_frag_T(&vt[(16 * n) * LDSS], LDSS);
      o[n] = wmma_f16(pa, vb, o[n]);
    }
    __syncthreads(); // protect vt/pb before next iteration rewrites them
  }

  // ---- normalize + store ctx (token-major, col = h*D + d) ----
  _Float16* op = ch + (size_t)(b * S_ + q0) * E_ + h * D_;
#pragma unroll
  for (int n = 0; n < 4; ++n) {
#pragma unroll
    for (int r = 0; r < 8; ++r) {
      int row = r + 8 * hi;
      float denom = (l[r] > 0.f) ? l[r] : 1.f;
      op[(size_t)row * E_ + 16 * n + rl] = (_Float16)(o[n][r] / denom);
    }
  }
}

// ---------------- host launcher ----------------
extern "C" void kernel_launch(void* const* d_in, const int* in_sizes, int n_in,
                              void* d_out, int out_size, void* d_ws, size_t ws_size,
                              hipStream_t stream) {
  (void)in_sizes; (void)n_in; (void)out_size; (void)ws_size;
  const float* x  = (const float*)d_in[0];
  const float* wq = (const float*)d_in[1];
  const float* bq = (const float*)d_in[2];
  const float* wk = (const float*)d_in[3];
  const float* bk = (const float*)d_in[4];
  const float* wv = (const float*)d_in[5];
  const float* bv = (const float*)d_in[6];
  const float* wo = (const float*)d_in[7];
  const float* bo = (const float*)d_in[8];
  float* out = (float*)d_out;

  char* ws = (char*)d_ws;
  size_t off = 0;
  auto alloc = [&](size_t bytes) {
    void* p = ws + off;
    off += (bytes + 255) & ~(size_t)255;
    return p;
  };
  _Float16* xh  = (_Float16*)alloc((size_t)BS_ * E_ * 2);
  _Float16* wqh = (_Float16*)alloc((size_t)E_ * E_ * 2);
  _Float16* wkh = (_Float16*)alloc((size_t)E_ * KVDIM_ * 2);
  _Float16* wvh = (_Float16*)alloc((size_t)E_ * KVDIM_ * 2);
  _Float16* woh = (_Float16*)alloc((size_t)E_ * E_ * 2);
  _Float16* qh  = (_Float16*)alloc((size_t)BS_ * E_ * 2);
  _Float16* kh  = (_Float16*)alloc((size_t)BS_ * KVDIM_ * 2);
  _Float16* vh  = (_Float16*)alloc((size_t)BS_ * KVDIM_ * 2);
  _Float16* chx = (_Float16*)alloc((size_t)BS_ * E_ * 2);

  const dim3 cvtg(1024), cvtb(256);
  cvt_f32_f16_kernel<<<cvtg, cvtb, 0, stream>>>(x,  xh,  BS_ * E_);
  cvt_f32_f16_kernel<<<cvtg, cvtb, 0, stream>>>(wq, wqh, E_ * E_);
  cvt_f32_f16_kernel<<<cvtg, cvtb, 0, stream>>>(wk, wkh, E_ * KVDIM_);
  cvt_f32_f16_kernel<<<cvtg, cvtb, 0, stream>>>(wv, wvh, E_ * KVDIM_);
  cvt_f32_f16_kernel<<<cvtg, cvtb, 0, stream>>>(wo, woh, E_ * E_);

  const dim3 gblk(256);
  gemm_wmma_kernel<true><<<dim3(BS_ / 128, E_ / 64),     gblk, 0, stream>>>(xh, wqh, bq, qh,  BS_, E_,     E_);
  gemm_wmma_kernel<true><<<dim3(BS_ / 128, KVDIM_ / 64), gblk, 0, stream>>>(xh, wkh, bk, kh,  BS_, KVDIM_, E_);
  gemm_wmma_kernel<true><<<dim3(BS_ / 128, KVDIM_ / 64), gblk, 0, stream>>>(xh, wvh, bv, vh,  BS_, KVDIM_, E_);

  attn_swa_kernel<<<B_ * H_ * (S_ / 16), 32, 0, stream>>>(qh, kh, vh, chx);

  gemm_wmma_kernel<false><<<dim3(BS_ / 128, E_ / 64),    gblk, 0, stream>>>(chx, woh, bo, out, BS_, E_,    E_);
}
